// QuantumLLM_33603824124324
// MI455X (gfx1250) — compile-verified, matrix operand order checked by reading, not run on hardware
//
#include <hip/hip_runtime.h>
#include <hip/hip_bf16.h>

// ---------------------------------------------------------------------------
// QuantumLLM forward for MI455X (gfx1250, wave32, WMMA).
//
// Roofline reasoning:
//  - Final GEMM [768x512]x[512x32000]: 65.5MB weight read + 98.3MB logits
//    write ~= 164MB -> ~7us @ 23.3TB/s. Compute (~25 GFLOP) is negligible on
//    bf16 WMMA, so weights stay f32 in memory and are converted to bf16 while
//    staging through LDS (coalesced float4 loads, 1KB tile shared by 4 waves,
//    each wave reusing it across 4 row-tiles -> 16x fewer weight transactions
//    than a naive per-wave strided load).
//  - Attention factorizes: cos(ph_l-ph_m) = cos*cos + sin*sin, so scores are
//    two [384x256]x[256x384] bf16 WMMA GEMMs; coh factorizes into column sums.
//  - All intermediates (<10MB) live in workspace / L2.
// ---------------------------------------------------------------------------

#define DEV __device__ __forceinline__

typedef __attribute__((ext_vector_type(16))) __bf16 v16bf;
typedef __attribute__((ext_vector_type(8)))  float  v8f;

static constexpr int Bb = 2, Ll = 384, Dd = 256, Vv = 32000, NLl = 3;
static constexpr int ROWS = Bb * Ll;          // 768
static constexpr float PIf  = 3.14159265358979323846f;
static constexpr float PHIf = 1.61803398874989484820f;

// ---------------- block reductions (LDS tree) ----------------
template<int NT>
DEV float block_sum(float v, float* sm) {
  int t = threadIdx.x;
  sm[t] = v; __syncthreads();
#pragma unroll
  for (int off = NT >> 1; off > 0; off >>= 1) {
    if (t < off) sm[t] += sm[t + off];
    __syncthreads();
  }
  float r = sm[0]; __syncthreads();
  return r;
}

template<int NT>
DEV float block_max(float v, float* sm) {
  int t = threadIdx.x;
  sm[t] = v; __syncthreads();
#pragma unroll
  for (int off = NT >> 1; off > 0; off >>= 1) {
    if (t < off) sm[t] = fmaxf(sm[t], sm[t + off]);
    __syncthreads();
  }
  float r = sm[0]; __syncthreads();
  return r;
}

// LayerNorm of one value within a row of NT values (one value per thread)
template<int NT>
DEV float ln_val(float v, float eps, float* sm) {
  const float invN = 1.0f / (float)NT;
  float mu  = block_sum<NT>(v, sm) * invN;
  float d   = v - mu;
  float var = block_sum<NT>(d * d, sm) * invN;
  return d * rsqrtf(var + eps);
}

DEV float clamp1(float v) { return fminf(1.0f, fmaxf(-1.0f, v)); }
DEV float sigmoidf(float v) { return 1.0f / (1.0f + __expf(-v)); }
DEV __bf16 f2bf(float f) { return (__bf16)f; }

// ---------------- WMMA tile loaders (CDNA5 16-bit layouts) ----------------
// A-matrix 16x32 bf16: lane&15 = row M; lanes<16 hold K {0..7,16..23},
// lanes>=16 hold K {8..15,24..31}. Elements 0..7 -> first K-octet, 8..15 ->
// second K-octet (+16).
DEV v16bf load_tileA_bf(const __bf16* src, int ld, int r0, int k0, int lane) {
  int m  = r0 + (lane & 15);
  int kb = k0 + ((lane & 16) ? 8 : 0);
  const __bf16* p = src + (long)m * ld + kb;
  v16bf a;
#pragma unroll
  for (int j = 0; j < 8; ++j) { a[j] = p[j]; a[j + 8] = p[16 + j]; }
  return a;
}

// B-matrix 32x16 bf16 (K x N): lane&15 = column N; same per-lane K mapping.
DEV v16bf load_tileB_bf(const __bf16* src, int ld, int k0, int n0, int lane) {
  int n  = n0 + (lane & 15);
  int kb = k0 + ((lane & 16) ? 8 : 0);
  v16bf b;
#pragma unroll
  for (int j = 0; j < 8; ++j) {
    b[j]     = src[(long)(kb + j) * ld + n];
    b[j + 8] = src[(long)(kb + 16 + j) * ld + n];
  }
  return b;
}

// C/D 16x16 f32: vgpr r -> row (r + 8*(lane>=16)), col = lane&15.
DEV void store_tileC(float* dst, long ld, int r0, int n0, int lane, v8f c,
                     float scale) {
  int n  = n0 + (lane & 15);
  int rb = r0 + ((lane & 16) ? 8 : 0);
#pragma unroll
  for (int r = 0; r < 8; ++r) dst[(long)(rb + r) * ld + n] = c[r] * scale;
}

DEV v8f wmma_bf16(v16bf a, v16bf b, v8f c) {
  return __builtin_amdgcn_wmma_f32_16x16x32_bf16(
      false, a, false, b, (short)0, c, false, false);
}

// ---------------- Stage 1: embedding + dynamic phase space ----------------
__global__ void k_embed_phase(const int* __restrict__ x,
                              const float* __restrict__ emb_real,
                              const float* __restrict__ layer_phases,
                              const float* __restrict__ iw,
                              const float* __restrict__ energy,
                              const float* __restrict__ excf,
                              float* __restrict__ real,
                              float* __restrict__ imag) {
  __shared__ float sm[256];
  int row = blockIdx.x;            // b*L + l
  int d   = threadIdx.x;
  long idx = (long)row * Dd + d;
  float re = emb_real[(long)x[row] * Dd + d] * 0.1f;
  float fr = 0.f, fi = 0.f;
#pragma unroll
  for (int i = 0; i < NLl; ++i) {
    float pa = tanhf(layer_phases[i * Dd + d]) * PIf * PHIf;
    float r  = ln_val<256>(re * __cosf(pa), 1e-8f, sm);
    float im = ln_val<256>(re * __sinf(pa), 1e-8f, sm);
    float w  = tanhf(iw[i * Dd + d]);
    r *= w; im *= w;
    float ang = 2.0f * PIf * (float)i / (float)NLl;
    float ca = __cosf(ang), sa = __sinf(ang);
    fr += r * ca - im * sa;
    fi += r * sa + im * ca;
  }
  float amp  = sqrtf(fr * fr + fi * fi + 1e-8f);
  float coll = (amp < 0.1f) ? 1.0f : 0.0f;
  float exc  = __expf(energy[d]) * excf[0];
  fr += coll * exc; fi += coll * exc;
  float inv = rsqrtf(fr * fr + fi * fi + 1e-8f);
  real[idx] = clamp1(fr * inv);
  imag[idx] = clamp1(fi * inv);
}

// ---------------- per-layer: row means of W_attn[i] ----------------
// one block per output d; coalesced row read + LDS reduce
__global__ void k_wmean(const float* __restrict__ Wi, float* __restrict__ wm) {
  __shared__ float sm[256];
  int d = blockIdx.x;
  float s = block_sum<256>(Wi[(long)d * Dd + threadIdx.x], sm);
  if (threadIdx.x == 0) wm[d] = s * (1.0f / (float)Dd);
}

// ---------------- per-layer: LN + phases + bf16 operand prep ----------------
__global__ void k_prep(const float* __restrict__ real,
                       const float* __restrict__ imag,
                       const float* __restrict__ wm,
                       float* __restrict__ cp, float* __restrict__ sp,
                       __bf16* __restrict__ cpb, __bf16* __restrict__ spb,
                       __bf16* __restrict__ wcpb, __bf16* __restrict__ wspb,
                       __bf16* __restrict__ nrb, __bf16* __restrict__ nib) {
  __shared__ float sm[256];
  int row = blockIdx.x;
  int d   = threadIdx.x;
  long idx = (long)row * Dd + d;
  float nr = ln_val<256>(real[idx], 1e-5f, sm);
  float ni = ln_val<256>(imag[idx], 1e-5f, sm);
  float ph = atan2f(ni + 1e-8f, nr + 1e-8f);
  float c = __cosf(ph), s = __sinf(ph);
  cp[idx] = c; sp[idx] = s;
  cpb[idx] = f2bf(c); spb[idx] = f2bf(s);
  float w = wm[d];
  wcpb[idx] = f2bf(w * c); wspb[idx] = f2bf(w * s);
  nrb[idx] = f2bf(nr); nib[idx] = f2bf(ni);
}

// ---------------- per-layer: column sums for coherence ----------------
__global__ void k_colsum(const float* __restrict__ cp,
                         const float* __restrict__ sp,
                         float* __restrict__ scos, float* __restrict__ ssin) {
  int tid = blockIdx.x * blockDim.x + threadIdx.x;   // b*D + d
  int b = tid >> 8, d = tid & 255;
  float sc = 0.f, ss = 0.f;
  const float* pc = cp + (long)b * Ll * Dd + d;
  const float* ps = sp + (long)b * Ll * Dd + d;
  for (int m = 0; m < Ll; ++m) { sc += pc[(long)m * Dd]; ss += ps[(long)m * Dd]; }
  scos[tid] = sc; ssin[tid] = ss;
}

// ---------------- per-layer: scores GEMM (WMMA bf16) ----------------
// scores[b] = (CP @ WCP^T + SP @ WSP^T) * D^-0.5   [L x L]
__global__ void k_scores(const __bf16* __restrict__ cpb,
                         const __bf16* __restrict__ spb,
                         const __bf16* __restrict__ wcpb,
                         const __bf16* __restrict__ wspb,
                         float* __restrict__ scores) {
  int lane = threadIdx.x & 31;
  int tile = blockIdx.x * 4 + (threadIdx.x >> 5);    // 2*24*24 = 1152 tiles
  int b  = tile / 576;
  int r  = (tile % 576) / 24;
  int cc = tile % 24;
  int r0 = r * 16, m0 = cc * 16;
  long base = (long)b * Ll * Dd;
  v8f acc = {};
  for (int k0 = 0; k0 < Dd; k0 += 32) {
    v16bf a1 = load_tileA_bf(cpb + base, Dd, r0, k0, lane);
    v16bf b1 = load_tileA_bf(wcpb + base, Dd, m0, k0, lane); // row of WCP = col of B
    acc = wmma_bf16(a1, b1, acc);
    v16bf a2 = load_tileA_bf(spb + base, Dd, r0, k0, lane);
    v16bf b2 = load_tileA_bf(wspb + base, Dd, m0, k0, lane);
    acc = wmma_bf16(a2, b2, acc);
  }
  store_tileC(scores + (long)b * Ll * Ll, Ll, r0, m0, lane, acc, 0.0625f);
}

// ---------------- per-layer: masked softmax + clip -> bf16 attn ----------------
__global__ void k_softmax(const float* __restrict__ scores,
                          const int* __restrict__ x,
                          __bf16* __restrict__ attnb) {
  __shared__ float sm[128];
  int row = blockIdx.x;            // b*L + l
  int b = row / Ll;
  const float* srow = scores + (long)row * Ll;
  float v[3];
  float mx = -INFINITY;
#pragma unroll
  for (int j = 0; j < 3; ++j) {
    int m = threadIdx.x + j * 128;
    float s = srow[m];
    if (x[b * Ll + m] == 0) s = -INFINITY;   // PAD mask on keys
    v[j] = s;
    mx = fmaxf(mx, s);
  }
  mx = block_max<128>(mx, sm);
  float lsum = 0.f;
#pragma unroll
  for (int j = 0; j < 3; ++j) { v[j] = __expf(v[j] - mx); lsum += v[j]; }
  float tot = block_sum<128>(lsum, sm);
  float inv = 1.0f / tot;
#pragma unroll
  for (int j = 0; j < 3; ++j) {
    int m = threadIdx.x + j * 128;
    float p = fminf(1.0f, fmaxf(1e-6f, v[j] * inv));
    attnb[(long)row * Ll + m] = f2bf(p);
  }
}

// ---------------- per-layer: attention output GEMMs (WMMA bf16) ----------------
// out_r = attn @ nr, out_i = attn @ ni   [L x D], K = L
__global__ void k_outgemm(const __bf16* __restrict__ attnb,
                          const __bf16* __restrict__ nrb,
                          const __bf16* __restrict__ nib,
                          float* __restrict__ out_r,
                          float* __restrict__ out_i) {
  int lane = threadIdx.x & 31;
  int tile = blockIdx.x * 4 + (threadIdx.x >> 5);    // 2*24*16 = 768 tiles
  int b   = tile / (24 * 16);
  int rem = tile % (24 * 16);
  int r0  = (rem / 16) * 16;
  int n0  = (rem % 16) * 16;
  const __bf16* A  = attnb + (long)b * Ll * Ll;
  const __bf16* Br = nrb + (long)b * Ll * Dd;
  const __bf16* Bi = nib + (long)b * Ll * Dd;
  v8f accr = {}, acci = {};
  for (int k0 = 0; k0 < Ll; k0 += 32) {
    v16bf a  = load_tileA_bf(A, Ll, r0, k0, lane);
    accr = wmma_bf16(a, load_tileB_bf(Br, Dd, k0, n0, lane), accr);
    acci = wmma_bf16(a, load_tileB_bf(Bi, Dd, k0, n0, lane), acci);
  }
  store_tileC(out_r + (long)b * Ll * Dd, Dd, r0, n0, lane, accr, 1.0f);
  store_tileC(out_i + (long)b * Ll * Dd, Dd, r0, n0, lane, acci, 1.0f);
}

// ---------------- per-layer: phase-preservation update ----------------
__global__ void k_post(const float* __restrict__ out_r,
                       const float* __restrict__ out_i,
                       const float* __restrict__ cp,
                       const float* __restrict__ sp,
                       const float* __restrict__ scos,
                       const float* __restrict__ ssin,
                       const float* __restrict__ cohf,   // scalar for layer i
                       const float* __restrict__ pp,     // [D] for layer i
                       float* __restrict__ real,
                       float* __restrict__ imag) {
  __shared__ float sm[256];
  int row = blockIdx.x;
  int b = row / Ll;
  int d = threadIdx.x;
  long idx = (long)row * Dd + d;
  float orv = out_r[idx], oiv = out_i[idx];
  float contrib = cp[idx] * scos[b * Dd + d] + sp[idx] * ssin[b * Dd + d];
  float coh = block_sum<256>(contrib, sm) * (1.0f / ((float)Ll * (float)Dd));
  float cf  = sigmoidf(cohf[0]) * coh;
  float phase = atan2f(oiv + 1e-8f, orv + 1e-8f);
  float pres  = sigmoidf(pp[d]) * cf;
  float pr = orv * __cosf(phase * pres);
  float t  = clamp1(ln_val<256>(real[idx] + 0.01f * pr, 1e-8f, sm));
  real[idx] = t;
  imag[idx] = t;    // reference: imag = clamp(real)
}

// ---------------- combine: LN over concat(real, imag) -> bf16 ----------------
__global__ void k_combine(const float* __restrict__ real,
                          const float* __restrict__ imag,
                          __bf16* __restrict__ comb) {
  __shared__ float sm[256];
  int row = blockIdx.x;
  int d = threadIdx.x;
  long idx = (long)row * Dd + d;
  float a = real[idx], bv = imag[idx];
  float mu = block_sum<256>(a + bv, sm) * (1.0f / 512.0f);
  float da = a - mu, db = bv - mu;
  float var = block_sum<256>(da * da + db * db, sm) * (1.0f / 512.0f);
  float inv = rsqrtf(var + 1e-5f);
  comb[(long)row * 512 + d]       = f2bf(da * inv);
  comb[(long)row * 512 + 256 + d] = f2bf(db * inv);
}

// ---------------- final logits GEMM (WMMA bf16, LDS-staged f32 weights) ----
// logits = clip((comb @ out_w + out_b) * 0.1, -10, 10)   [768 x 32000]
// Block = 4 waves. Block owns one 16-col slab; the 32x16 f32 weight tile is
// loaded ONCE per k-step with coalesced float4 loads, converted to bf16 into
// a 1KB LDS tile, and shared by all 4 waves; each wave reuses it across 4
// row-tiles (M=64). Weight transactions per WMMA drop 16x vs naive.
__global__ void k_logits(const __bf16* __restrict__ comb,
                         const float* __restrict__ out_w,
                         const float* __restrict__ out_b,
                         float* __restrict__ out) {
  __shared__ __bf16 bt[32 * 16];           // K x N bf16 tile
  int tid  = threadIdx.x;
  int lane = tid & 31;
  int wave = tid >> 5;
  int n0 = (blockIdx.x % 2000) * 16;
  int rg = (blockIdx.x / 2000);            // 0..2, 256 rows each
  int rbase = rg * 256 + wave * 64;        // this wave: rows [rbase, rbase+64)
  v8f acc[4] = {};
  for (int k0 = 0; k0 < 512; k0 += 32) {
    __syncthreads();                       // previous tile fully consumed
    {
      // cooperative coalesced load: thread -> 4 consecutive floats of tile
      int kk = tid >> 2;                   // 0..31
      int nn = (tid & 3) * 4;              // 0,4,8,12
      const float4 w4 = *(const float4*)(out_w + (long)(k0 + kk) * Vv + n0 + nn);
      __bf16* dst = &bt[kk * 16 + nn];
      dst[0] = f2bf(w4.x); dst[1] = f2bf(w4.y);
      dst[2] = f2bf(w4.z); dst[3] = f2bf(w4.w);
    }
    __syncthreads();                       // tile visible to all waves
    // read B operand from LDS in WMMA B layout
    v16bf b;
    {
      int n  = lane & 15;
      int kb = (lane & 16) ? 8 : 0;
#pragma unroll
      for (int j = 0; j < 8; ++j) {
        b[j]     = bt[(kb + j) * 16 + n];
        b[j + 8] = bt[(kb + 16 + j) * 16 + n];
      }
    }
#pragma unroll
    for (int r = 0; r < 4; ++r) {
      v16bf a = load_tileA_bf(comb, 512, rbase + r * 16, k0, lane);
      acc[r] = wmma_bf16(a, b, acc[r]);
    }
  }
  int n  = n0 + (lane & 15);
  float bias = out_b[n];
#pragma unroll
  for (int r = 0; r < 4; ++r) {
    int rb = rbase + r * 16 + ((lane & 16) ? 8 : 0);
#pragma unroll
    for (int rr = 0; rr < 8; ++rr) {
      float v = (acc[r][rr] + bias) * 0.1f;
      out[(long)(rb + rr) * Vv + n] = fminf(10.0f, fmaxf(-10.0f, v));
    }
  }
}

// ---------------------------------------------------------------------------
extern "C" void kernel_launch(void* const* d_in, const int* in_sizes, int n_in,
                              void* d_out, int out_size, void* d_ws, size_t ws_size,
                              hipStream_t stream) {
  const int*   x      = (const int*)  d_in[0];
  const float* emb_r  = (const float*)d_in[1];
  // d_in[2] emb_imag: unused downstream (matches reference)
  const float* lph    = (const float*)d_in[3];
  const float* iw     = (const float*)d_in[4];
  const float* energy = (const float*)d_in[5];
  const float* excf   = (const float*)d_in[6];
  const float* pp     = (const float*)d_in[7];
  const float* cohf   = (const float*)d_in[8];
  const float* Wattn  = (const float*)d_in[9];
  const float* out_w  = (const float*)d_in[10];
  const float* out_b  = (const float*)d_in[11];
  float* out = (float*)d_out;

  // workspace carve-out (256B aligned slices)
  char* base = (char*)d_ws;
  size_t off = 0;
  auto alloc = [&](size_t bytes) -> void* {
    void* p = base + off;
    off = (off + bytes + 255) & ~(size_t)255;
    return p;
  };
  const size_t fRD = (size_t)ROWS * Dd * sizeof(float);
  const size_t hRD = (size_t)ROWS * Dd * sizeof(__bf16);
  float*  real  = (float*) alloc(fRD);
  float*  imag  = (float*) alloc(fRD);
  float*  cp    = (float*) alloc(fRD);
  float*  sp    = (float*) alloc(fRD);
  float*  out_r = (float*) alloc(fRD);
  float*  out_i = (float*) alloc(fRD);
  __bf16* cpb   = (__bf16*)alloc(hRD);
  __bf16* spb   = (__bf16*)alloc(hRD);
  __bf16* wcpb  = (__bf16*)alloc(hRD);
  __bf16* wspb  = (__bf16*)alloc(hRD);
  __bf16* nrb   = (__bf16*)alloc(hRD);
  __bf16* nib   = (__bf16*)alloc(hRD);
  float*  wm    = (float*) alloc(Dd * sizeof(float));
  float*  scos  = (float*) alloc((size_t)Bb * Dd * sizeof(float));
  float*  ssin  = (float*) alloc((size_t)Bb * Dd * sizeof(float));
  float*  score = (float*) alloc((size_t)Bb * Ll * Ll * sizeof(float));
  __bf16* attnb = (__bf16*)alloc((size_t)Bb * Ll * Ll * sizeof(__bf16));
  __bf16* comb  = (__bf16*)alloc((size_t)ROWS * 2 * Dd * sizeof(__bf16));
  (void)ws_size; (void)in_sizes; (void)n_in; (void)out_size;

  // Stage 1: embedding + phase space
  k_embed_phase<<<ROWS, 256, 0, stream>>>(x, emb_r, lph, iw, energy, excf,
                                          real, imag);

  // Stage 2: three attention layers
  for (int i = 0; i < NLl; ++i) {
    k_wmean  <<<Dd, 256, 0, stream>>>(Wattn + (long)i * Dd * Dd, wm);
    k_prep   <<<ROWS, 256, 0, stream>>>(real, imag, wm, cp, sp,
                                        cpb, spb, wcpb, wspb, nrb, nib);
    k_colsum <<<(Bb * Dd) / 256, 256, 0, stream>>>(cp, sp, scos, ssin);
    k_scores <<<(Bb * 24 * 24) / 4, 128, 0, stream>>>(cpb, spb, wcpb, wspb,
                                                      score);
    k_softmax<<<ROWS, 128, 0, stream>>>(score, x, attnb);
    k_outgemm<<<(Bb * 24 * 16) / 4, 128, 0, stream>>>(attnb, nrb, nib,
                                                      out_r, out_i);
    k_post   <<<ROWS, 256, 0, stream>>>(out_r, out_i, cp, sp, scos, ssin,
                                        cohf + i, pp + (long)i * Dd,
                                        real, imag);
  }

  // Stage 3: combine + logits
  k_combine<<<ROWS, 256, 0, stream>>>(real, imag, comb);
  k_logits <<<3 * 2000, 128, 0, stream>>>(comb, out_w, out_b, out);
}
